// CoulombNet_9414568313024
// MI455X (gfx1250) — matrix operand order repaired and checked by reading, not compiled, not run on hardware
//
#include <hip/hip_runtime.h>
#include <hip/hip_bf16.h>

// ---------------------------------------------------------------------------
// Hierarchical GATv2 + TopK pooling forward, CDNA5 (gfx1250, wave32).
// Dense projections: v_wmma_f32_16x16x32_bf16 (fp32 accum). xl/xr are fused
// into one dual-output WMMA kernel (A fragment built once, two B tiles staged
// in LDS). W tiles staged to LDS as packed bf16 (RNE) once per block; A path
// uses truncating f32->bf16 pack (2 VALU/pair). Edge passes are streaming
// kernels with f32 atomics; segment-max uses order-preserving u32 keys.
//
// Input layout assumption (setup_inputs() insertion order, nested dicts
// flattened in insertion order):
//   0:x[50000,64] 1:edge_attr[800000,8]
//   2..13  layer0: Wl,Wr,We,att,bc,Wh,bh,gamma,beta,rm,rv,p
//   14..25 layer1, 26..37 layer2
//   38:Wf1[192,64] 39:bf1[64] 40:Wf2[64,1] 41:bf2[1]
//   42:edge_index[2,800000] (int32) 43:batch[50000] (int32)
// Output: 256 f32.  Workspace required: ~118 MB.
// ---------------------------------------------------------------------------

#define N_NODES   50000
#define N_PAD     50016
#define N_EDGES   800000
#define N_TOT     (N_EDGES + N_NODES)   // edge slots + self-loop slots
#define NUM_GRAPHS 256
#define HID       64
#define EDGE_DIM  8
#define HEADS     2
#define FDIM      128                    // HEADS*HID
#define BN_EPS    1e-5f
#define NEG_SLOPE 0.2f
#define FMIN      (-3.402823466e38f)

typedef __attribute__((ext_vector_type(16))) __bf16 v16bf;
typedef __attribute__((ext_vector_type(8)))  float  v8f;

// round-to-nearest-even pack (cold path: weight staging, once per block)
__device__ __forceinline__ unsigned short f2bf_rne(float f) {
  unsigned u = __float_as_uint(f);
  u += 0x7FFFu + ((u >> 16) & 1u);
  return (unsigned short)(u >> 16);
}
__device__ __forceinline__ unsigned pkbf_rne(float a, float b) {
  return (unsigned)f2bf_rne(a) | ((unsigned)f2bf_rne(b) << 16);
}
// truncating pack (hot path: per-wave A fragments) -> 2 VALU ops
__device__ __forceinline__ unsigned pkbf_tr(float a, float b) {
  return (__float_as_uint(b) & 0xFFFF0000u) | (__float_as_uint(a) >> 16);
}
struct BFr { unsigned u[8]; };           // 16 packed bf16 = one WMMA operand

__device__ __forceinline__ unsigned f_to_key(float f) {
  unsigned u = __float_as_uint(f);
  return (u & 0x80000000u) ? ~u : (u | 0x80000000u);
}
__device__ __forceinline__ float key_to_f(unsigned k) {
  unsigned u = (k & 0x80000000u) ? (k ^ 0x80000000u) : ~k;
  return __uint_as_float(u);
}

// Build A fragment for one 32-wide K chunk (contiguous k order, same as load_b_frag):
// lane row = row0+mn; lanes 0-15: K = k0+{0..15};
// lanes 16-31: K = k0+{16..31}.  4x global_load_b128.
__device__ __forceinline__ BFr load_a_frag(const float* Achunk, int hi) {
  const float4* Ar = (const float4*)Achunk;
  float4 q0 = Ar[hi * 4 + 0];
  float4 q1 = Ar[hi * 4 + 1];
  float4 q2 = Ar[hi * 4 + 2];
  float4 q3 = Ar[hi * 4 + 3];
  BFr ab;
  ab.u[0] = pkbf_tr(q0.x, q0.y); ab.u[1] = pkbf_tr(q0.z, q0.w);
  ab.u[2] = pkbf_tr(q1.x, q1.y); ab.u[3] = pkbf_tr(q1.z, q1.w);
  ab.u[4] = pkbf_tr(q2.x, q2.y); ab.u[5] = pkbf_tr(q2.z, q2.w);
  ab.u[6] = pkbf_tr(q3.x, q3.y); ab.u[7] = pkbf_tr(q3.z, q3.w);
  return ab;
}
// B fragment: 8 contiguous packed uints in LDS -> 2x ds_load_b128
__device__ __forceinline__ BFr load_b_frag(const unsigned* sWp, int KT2,
                                           int mn, int k0, int hi) {
  const uint4* bp = (const uint4*)&sWp[mn * KT2 + (k0 + hi * 16) / 2];
  uint4 b0 = bp[0], b1 = bp[1];
  BFr bb;
  bb.u[0] = b0.x; bb.u[1] = b0.y; bb.u[2] = b0.z; bb.u[3] = b0.w;
  bb.u[4] = b1.x; bb.u[5] = b1.y; bb.u[6] = b1.z; bb.u[7] = b1.w;
  return bb;
}

// ---------------- generic fill ----------------
__global__ void fill_u32_k(unsigned* __restrict__ p, unsigned v, int n) {
  int i = blockIdx.x * blockDim.x + threadIdx.x;
  if (i < n) p[i] = v;
}

// ---------------- graph bookkeeping ----------------
__global__ void graph_count_k(const int* __restrict__ batch, int* __restrict__ counts) {
  int n = blockIdx.x * blockDim.x + threadIdx.x;
  if (n < N_NODES) atomicAdd(&counts[batch[n]], 1);
}
__global__ void graph_start_k(const int* __restrict__ counts, int* __restrict__ starts) {
  if (threadIdx.x == 0 && blockIdx.x == 0) {
    int s = 0;
    for (int g = 0; g < NUM_GRAPHS; ++g) { starts[g] = s; s += counts[g]; }
  }
}

// ---- single-output WMMA GEMM: C = A[M,KT] @ W[KT,N] (+bias, relu) ---------
// KT compile-time -> fully unrolled v_wmma chain. M % 16 == 0 at call sites.
template <int KT, bool ADD_BIAS, bool RELU>
__global__ void gemm_wmma_bf16(const float* __restrict__ A, const float* __restrict__ W,
                               const float* __restrict__ bias, float* __restrict__ C,
                               int M, int N) {
  __shared__ unsigned sWp[16 * (KT / 2)];  // [col][k/2] packed bf16 pairs
  const int tid  = threadIdx.x;
  const int col0 = blockIdx.y * 16;
  for (int idx = tid; idx < 16 * (KT / 2); idx += blockDim.x) {
    int col = idx & 15;
    int k2  = idx >> 4;
    sWp[col * (KT / 2) + k2] = pkbf_rne(W[(size_t)(2 * k2) * N + col0 + col],
                                        W[(size_t)(2 * k2 + 1) * N + col0 + col]);
  }
  __syncthreads();

  const int lane = tid & 31;
  const int wave = tid >> 5;
  const int row0 = (blockIdx.x * (int)(blockDim.x >> 5) + wave) * 16;
  if (row0 >= M) return;                    // wave-uniform; EXEC stays all-1s
  const int hi = lane >> 4;
  const int mn = lane & 15;

  const float* Arow = A + (size_t)(row0 + mn) * KT;
  __builtin_prefetch(Arow, 0, 3);           // global_prefetch_b8

  v8f acc = {};
#pragma unroll
  for (int k0 = 0; k0 < KT; k0 += 32) {
    BFr ab = load_a_frag(Arow + k0, hi);
    BFr bb = load_b_frag(sWp, KT / 2, mn, k0, hi);
    acc = __builtin_amdgcn_wmma_f32_16x16x32_bf16(
        false, __builtin_bit_cast(v16bf, ab),
        false, __builtin_bit_cast(v16bf, bb), (short)0, acc, false, false);
  }
  // C/D layout: VGPR r -> M = row0 + hi*8 + r, N = col0 + mn.
  float bv = ADD_BIAS ? bias[col0 + mn] : 0.f;
#pragma unroll
  for (int r = 0; r < 8; ++r) {
    float v = acc[r];
    if (ADD_BIAS) v += bv;
    if (RELU) v = v > 0.f ? v : 0.f;
    C[(size_t)(row0 + hi * 8 + r) * N + col0 + mn] = v;
  }
}

// ---- dual-output WMMA GEMM: C0 = A@W0, C1 = A@W1 (shared A fragments) -----
// Used for xl/xr: halves A traffic and f32->bf16 conversion work.
template <int KT>
__global__ void gemm2_wmma_bf16(const float* __restrict__ A,
                                const float* __restrict__ W0, const float* __restrict__ W1,
                                float* __restrict__ C0, float* __restrict__ C1,
                                int M, int N) {
  const int TL = 16 * (KT / 2);
  __shared__ unsigned sW0[16 * (KT / 2)];
  __shared__ unsigned sW1[16 * (KT / 2)];
  const int tid  = threadIdx.x;
  const int col0 = blockIdx.y * 16;
  for (int idx = tid; idx < 2 * TL; idx += blockDim.x) {
    int which = idx >= TL;
    int i   = which ? idx - TL : idx;
    int col = i & 15;
    int k2  = i >> 4;
    const float* Ws = which ? W1 : W0;
    unsigned v = pkbf_rne(Ws[(size_t)(2 * k2) * N + col0 + col],
                          Ws[(size_t)(2 * k2 + 1) * N + col0 + col]);
    (which ? sW1 : sW0)[col * (KT / 2) + k2] = v;
  }
  __syncthreads();

  const int lane = tid & 31;
  const int wave = tid >> 5;
  const int row0 = (blockIdx.x * (int)(blockDim.x >> 5) + wave) * 16;
  if (row0 >= M) return;
  const int hi = lane >> 4;
  const int mn = lane & 15;

  const float* Arow = A + (size_t)(row0 + mn) * KT;
  __builtin_prefetch(Arow, 0, 3);

  v8f acc0 = {}, acc1 = {};
#pragma unroll
  for (int k0 = 0; k0 < KT; k0 += 32) {
    BFr ab = load_a_frag(Arow + k0, hi);
    v16bf av = __builtin_bit_cast(v16bf, ab);
    BFr b0 = load_b_frag(sW0, KT / 2, mn, k0, hi);
    BFr b1 = load_b_frag(sW1, KT / 2, mn, k0, hi);
    acc0 = __builtin_amdgcn_wmma_f32_16x16x32_bf16(
        false, av, false, __builtin_bit_cast(v16bf, b0), (short)0, acc0, false, false);
    acc1 = __builtin_amdgcn_wmma_f32_16x16x32_bf16(
        false, av, false, __builtin_bit_cast(v16bf, b1), (short)0, acc1, false, false);
  }
#pragma unroll
  for (int r = 0; r < 8; ++r) {
    size_t o = (size_t)(row0 + hi * 8 + r) * N + col0 + mn;
    C0[o] = acc0[r];
    C1[o] = acc1[r];
  }
}

// ---------------- self-loop edge_attr mean ----------------
__global__ void eattr_sum_k(const int* __restrict__ edst, const float* __restrict__ eattr,
                            const int* __restrict__ emask,
                            float* __restrict__ asum, float* __restrict__ acnt) {
  int e = blockIdx.x * blockDim.x + threadIdx.x;
  if (e >= N_EDGES || !emask[e]) return;
  int d = edst[e];
#pragma unroll
  for (int k = 0; k < EDGE_DIM; ++k)
    atomicAdd(&asum[(size_t)d * EDGE_DIM + k], eattr[(size_t)e * EDGE_DIM + k]);
  atomicAdd(&acnt[d], 1.0f);
}
__global__ void self_attr_k(const float* __restrict__ asum, const float* __restrict__ acnt,
                            float* __restrict__ selfat) {
  int n = blockIdx.x * blockDim.x + threadIdx.x;
  if (n >= N_NODES) return;
  float c = fmaxf(acnt[n], 1.0f);
#pragma unroll
  for (int k = 0; k < EDGE_DIM; ++k)
    selfat[(size_t)n * EDGE_DIM + k] = asum[(size_t)n * EDGE_DIM + k] / c;
}

// ---------------- attention logits (edge proj fused) + segment max ---------
__global__ void logit_k(const int* __restrict__ esrc, const int* __restrict__ edst,
                        const float* __restrict__ eattr, const int* __restrict__ emask,
                        const int* __restrict__ nmask, const float* __restrict__ selfat,
                        const float* __restrict__ xl, const float* __restrict__ xr,
                        const float* __restrict__ We, const float* __restrict__ att,
                        float* __restrict__ logit, unsigned* __restrict__ mx) {
  int idx = blockIdx.x * blockDim.x + threadIdx.x;
  if (idx >= N_TOT * HEADS) return;
  int r = idx >> 1, h = idx & 1;
  int src, dst, valid;
  const float* a;
  if (r < N_EDGES) { src = esrc[r]; dst = edst[r]; valid = emask[r];
                     a = eattr + (size_t)r * EDGE_DIM; }
  else             { int n = r - N_EDGES; src = n; dst = n; valid = nmask[n];
                     a = selfat + (size_t)n * EDGE_DIM; }
  float av[EDGE_DIM];
#pragma unroll
  for (int k = 0; k < EDGE_DIM; ++k) av[k] = a[k];
  const float* xls = xl + (size_t)src * FDIM + h * HID;
  const float* xrd = xr + (size_t)dst * FDIM + h * HID;
  const float* Wec = We + h * HID;
  const float* atc = att + h * HID;
  float acc = 0.f;
  for (int d = 0; d < HID; ++d) {
    float ea = 0.f;
#pragma unroll
    for (int k = 0; k < EDGE_DIM; ++k) ea += av[k] * Wec[k * FDIM + d];
    float e = xls[d] + xrd[d] + ea;
    float l = e > 0.f ? e : NEG_SLOPE * e;
    acc += l * atc[d];
  }
  if (!valid) acc = FMIN;
  logit[(size_t)r * 2 + h] = acc;
  atomicMax(&mx[(size_t)dst * 2 + h], f_to_key(acc));
}

// ---------------- exp(logit - max) + segment sum ----------------
__global__ void expv_k(const int* __restrict__ edst, const int* __restrict__ emask,
                       const int* __restrict__ nmask,
                       float* __restrict__ logit, const unsigned* __restrict__ mx,
                       float* __restrict__ den) {
  int idx = blockIdx.x * blockDim.x + threadIdx.x;
  if (idx >= N_TOT * HEADS) return;
  int r = idx >> 1, h = idx & 1;
  int dst, valid;
  if (r < N_EDGES) { dst = edst[r]; valid = emask[r]; }
  else             { dst = r - N_EDGES; valid = nmask[dst]; }
  float ev = 0.f;
  if (valid) {
    float m = key_to_f(mx[(size_t)dst * 2 + h]);
    ev = __expf(logit[(size_t)r * 2 + h] - m);
  }
  logit[(size_t)r * 2 + h] = ev;              // overwrite logits with expv
  if (ev != 0.f) atomicAdd(&den[(size_t)dst * 2 + h], ev);
}

// ---------------- alpha-weighted message aggregation ----------------
__global__ void agg_k(const int* __restrict__ esrc, const int* __restrict__ edst,
                      const float* __restrict__ expv, const float* __restrict__ den,
                      const float* __restrict__ xl, float* __restrict__ out) {
  int idx = blockIdx.x * blockDim.x + threadIdx.x;
  if (idx >= N_TOT * FDIM / 2) return;        // 2 channels per thread
  int r = idx >> 6;                            // row slot
  int c0 = (idx & 63) * 2;                     // channel pair
  int src, dst;
  if (r < N_EDGES) { src = esrc[r]; dst = edst[r]; }
  else             { src = dst = r - N_EDGES; }
#pragma unroll
  for (int t = 0; t < 2; ++t) {
    int c = c0 + t, h = c >> 6;
    float ev = expv[(size_t)r * 2 + h];
    if (ev != 0.f) {
      float alpha = ev / fmaxf(den[(size_t)dst * 2 + h], 1e-16f);
      atomicAdd(&out[(size_t)dst * FDIM + c], alpha * xl[(size_t)src * FDIM + c]);
    }
  }
}

__global__ void add_bc_k(float* __restrict__ out, const float* __restrict__ bc) {
  int idx = blockIdx.x * blockDim.x + threadIdx.x;
  if (idx < N_NODES * FDIM) out[idx] += bc[idx & (FDIM - 1)];
}

// ---------------- BatchNorm (eval) ----------------
__global__ void bn_k(float* __restrict__ h, const float* __restrict__ gamma,
                     const float* __restrict__ beta, const float* __restrict__ rm,
                     const float* __restrict__ rv) {
  int idx = blockIdx.x * blockDim.x + threadIdx.x;
  if (idx >= N_NODES * HID) return;
  int d = idx & (HID - 1);
  h[idx] = (h[idx] - rm[d]) * rsqrtf(rv[d] + BN_EPS) * gamma[d] + beta[d];
}

// ---------------- TopK pooling ----------------
__global__ void pnorm_k(const float* __restrict__ p, float* __restrict__ out) {
  __shared__ float sh[64];
  int t = threadIdx.x;
  sh[t] = p[t] * p[t];
  __syncthreads();
  for (int s = 32; s > 0; s >>= 1) { if (t < s) sh[t] += sh[t + s]; __syncthreads(); }
  if (t == 0) out[0] = sqrtf(sh[0]);
}
__global__ void score_k(const float* __restrict__ h, const float* __restrict__ p,
                        const float* __restrict__ pnorm, const int* __restrict__ nmask,
                        const int* __restrict__ batch, int* __restrict__ vcnt,
                        float* __restrict__ score) {
  int n = blockIdx.x * blockDim.x + threadIdx.x;
  if (n >= N_NODES) return;
  float acc = 0.f;
  const float* hr = h + (size_t)n * HID;
#pragma unroll 8
  for (int d = 0; d < HID; ++d) acc += hr[d] * p[d];
  score[n] = acc / pnorm[0];
  if (nmask[n]) atomicAdd(&vcnt[batch[n]], 1);
}
__global__ void kcount_k(const int* __restrict__ vcnt, int* __restrict__ kk, float ratio) {
  int g = blockIdx.x * blockDim.x + threadIdx.x;
  if (g < NUM_GRAPHS) kk[g] = (int)ceilf(ratio * (float)vcnt[g]);
}
__global__ void topk_keep_k(const float* __restrict__ score, const int* __restrict__ nmask,
                            const int* __restrict__ batch, const int* __restrict__ starts,
                            const int* __restrict__ counts, const int* __restrict__ kk,
                            int* __restrict__ keep) {
  int n = blockIdx.x * blockDim.x + threadIdx.x;
  if (n >= N_NODES) return;
  if (!nmask[n]) { keep[n] = 0; return; }
  int g = batch[n];
  float si = score[n];
  int s0 = starts[g], c = counts[g], rank = 0;
  for (int j = s0; j < s0 + c; ++j) {        // stable descending order rank
    if (!nmask[j]) continue;
    float sj = score[j];
    if (sj > si || (sj == si && j < n)) rank++;
  }
  keep[n] = (rank < kk[g]) ? 1 : 0;
}
__global__ void scale_k(float* __restrict__ h, const float* __restrict__ score,
                        const int* __restrict__ keep) {
  int idx = blockIdx.x * blockDim.x + threadIdx.x;
  if (idx >= N_NODES * HID) return;
  int n = idx >> 6;
  h[idx] *= tanhf(score[n]) * (keep[n] ? 1.f : 0.f);
}
__global__ void emask_k(int* __restrict__ emask, const int* __restrict__ esrc,
                        const int* __restrict__ edst, const int* __restrict__ keep) {
  int e = blockIdx.x * blockDim.x + threadIdx.x;
  if (e < N_EDGES) emask[e] = emask[e] && keep[esrc[e]] && keep[edst[e]];
}

// ---------------- mean pool readout ----------------
__global__ void pool_sum_k(const float* __restrict__ h, const int* __restrict__ batch,
                           float* __restrict__ reads, int col0) {
  int idx = blockIdx.x * blockDim.x + threadIdx.x;
  if (idx >= N_NODES * HID) return;
  int n = idx >> 6, d = idx & (HID - 1);
  float v = h[idx];
  if (v != 0.f) atomicAdd(&reads[(size_t)batch[n] * (3 * HID) + col0 + d], v);
}
__global__ void pool_cnt_k(const int* __restrict__ keep, const int* __restrict__ batch,
                           int* __restrict__ pcnt) {
  int n = blockIdx.x * blockDim.x + threadIdx.x;
  if (n < N_NODES && keep[n]) atomicAdd(&pcnt[batch[n]], 1);
}
__global__ void pool_div_k(float* __restrict__ reads, const int* __restrict__ pcnt, int col0) {
  int idx = blockIdx.x * blockDim.x + threadIdx.x;
  if (idx >= NUM_GRAPHS * HID) return;
  int g = idx >> 6, d = idx & (HID - 1);
  reads[(size_t)g * (3 * HID) + col0 + d] /= fmaxf((float)pcnt[g], 1.f);
}

// ---------------- final MLP ----------------
__global__ void mlp1_k(const float* __restrict__ reads, const float* __restrict__ Wf1,
                       const float* __restrict__ bf1, float* __restrict__ z1) {
  int idx = blockIdx.x * blockDim.x + threadIdx.x;
  if (idx >= NUM_GRAPHS * HID) return;
  int g = idx >> 6, j = idx & (HID - 1);
  float acc = bf1[j];
  for (int t = 0; t < 3 * HID; ++t) acc += reads[(size_t)g * (3 * HID) + t] * Wf1[t * HID + j];
  z1[idx] = fmaxf(acc, 0.f);
}
__global__ void mlp2_k(const float* __restrict__ z1, const float* __restrict__ Wf2,
                       const float* __restrict__ bf2, float* __restrict__ out) {
  int g = blockIdx.x * blockDim.x + threadIdx.x;
  if (g >= NUM_GRAPHS) return;
  float acc = bf2[0];
#pragma unroll 8
  for (int j = 0; j < HID; ++j) acc += z1[(size_t)g * HID + j] * Wf2[j];
  out[g] = acc;
}

// ---------------------------------------------------------------------------
static inline unsigned cdiv(long n, int b) { return (unsigned)((n + b - 1) / b); }

extern "C" void kernel_launch(void* const* d_in, const int* in_sizes, int n_in,
                              void* d_out, int out_size, void* d_ws, size_t ws_size,
                              hipStream_t stream) {
  (void)in_sizes; (void)n_in; (void)out_size; (void)ws_size;
  const float* x      = (const float*)d_in[0];
  const float* eattr  = (const float*)d_in[1];
  const float* Wf1    = (const float*)d_in[38];
  const float* bf1    = (const float*)d_in[39];
  const float* Wf2    = (const float*)d_in[40];
  const float* bf2    = (const float*)d_in[41];
  const int*   eidx   = (const int*)d_in[42];
  const int*   batch  = (const int*)d_in[43];
  const int*   esrc   = eidx;
  const int*   edst   = eidx + N_EDGES;
  float*       outp   = (float*)d_out;

  // ---- workspace bump allocator ----
  char* base = (char*)d_ws;
  size_t off = 0;
  auto alloc = [&](size_t bytes) -> void* {
    off = (off + 255) & ~(size_t)255;
    void* p = base + off;
    off += bytes;
    return p;
  };
  float* h0     = (float*)alloc((size_t)N_PAD * HID * 4);
  float* h1     = (float*)alloc((size_t)N_PAD * HID * 4);
  float* xl     = (float*)alloc((size_t)N_PAD * FDIM * 4);
  float* xr     = (float*)alloc((size_t)N_PAD * FDIM * 4);
  float* gout   = (float*)alloc((size_t)N_PAD * FDIM * 4);
  float* selfat = (float*)alloc((size_t)N_PAD * EDGE_DIM * 4);
  float* asum   = (float*)alloc((size_t)N_PAD * EDGE_DIM * 4);
  float* acnt   = (float*)alloc((size_t)N_PAD * 4);
  float* logit  = (float*)alloc((size_t)N_TOT * 2 * 4);
  unsigned* mx  = (unsigned*)alloc((size_t)N_PAD * 2 * 4);
  float* den    = (float*)alloc((size_t)N_PAD * 2 * 4);
  float* score  = (float*)alloc((size_t)N_PAD * 4);
  float* pnorm  = (float*)alloc(256);
  int* maskA    = (int*)alloc((size_t)N_PAD * 4);
  int* maskB    = (int*)alloc((size_t)N_PAD * 4);
  int* emask    = (int*)alloc((size_t)N_EDGES * 4);
  int* counts   = (int*)alloc(NUM_GRAPHS * 4);
  int* starts   = (int*)alloc(NUM_GRAPHS * 4);
  int* vcnt     = (int*)alloc(NUM_GRAPHS * 4);
  int* kk       = (int*)alloc(NUM_GRAPHS * 4);
  int* pcnt     = (int*)alloc(NUM_GRAPHS * 4);
  float* reads  = (float*)alloc((size_t)NUM_GRAPHS * 3 * HID * 4);
  float* z1     = (float*)alloc((size_t)NUM_GRAPHS * HID * 4);

  const int B = 256;
  auto fill = [&](void* p, unsigned v, long n) {
    fill_u32_k<<<cdiv(n, B), B, 0, stream>>>((unsigned*)p, v, (int)n);
  };

  // ---- once-per-call init ----
  fill(emask, 1u, N_EDGES);
  fill(maskA, 1u, N_NODES);
  fill(counts, 0u, NUM_GRAPHS);
  fill(reads, 0u, NUM_GRAPHS * 3 * HID);
  graph_count_k<<<cdiv(N_NODES, B), B, 0, stream>>>(batch, counts);
  graph_start_k<<<1, 1, 0, stream>>>(counts, starts);

  const float ratios[3] = {0.9f, 0.7f, 0.5f};
  const float* hin = x;
  int* nmask = maskA;
  int* keep  = maskB;

  for (int L = 0; L < 3; ++L) {
    const float** P = (const float**)(d_in + 2 + L * 12);
    const float *Wl = P[0], *Wr = P[1], *We = P[2], *att = P[3], *bc = P[4],
                *Wh = P[5], *bh = P[6], *gamma = P[7], *beta = P[8],
                *rm = P[9], *rv = P[10], *pv = P[11];
    float* hn = (L & 1) ? h1 : h0;

    // zero per-layer accumulators
    fill(asum, 0u, (long)N_NODES * EDGE_DIM);
    fill(acnt, 0u, N_NODES);
    fill(mx,   0u, (long)N_NODES * 2);       // u32 0 == minimal ordered key
    fill(den,  0u, (long)N_NODES * 2);
    fill(gout, 0u, (long)N_NODES * FDIM);
    fill(vcnt, 0u, NUM_GRAPHS);
    fill(pcnt, 0u, NUM_GRAPHS);

    // node projections: xl = h@Wl, xr = h@Wr in one fused dual-WMMA pass
    {
      dim3 grid(cdiv(N_NODES / 16, 8), FDIM / 16);
      gemm2_wmma_bf16<HID><<<grid, 256, 0, stream>>>(hin, Wl, Wr, xl, xr,
                                                     N_NODES, FDIM);
    }
    // self-loop attr = mean of valid incoming edge_attr
    eattr_sum_k<<<cdiv(N_EDGES, B), B, 0, stream>>>(edst, eattr, emask, asum, acnt);
    self_attr_k<<<cdiv(N_NODES, B), B, 0, stream>>>(asum, acnt, selfat);
    // attention logits + segment max
    logit_k<<<cdiv((long)N_TOT * HEADS, B), B, 0, stream>>>(
        esrc, edst, eattr, emask, nmask, selfat, xl, xr, We, att, logit, mx);
    // softmax numerator + denominator
    expv_k<<<cdiv((long)N_TOT * HEADS, B), B, 0, stream>>>(edst, emask, nmask,
                                                           logit, mx, den);
    // weighted aggregation
    agg_k<<<cdiv((long)N_TOT * FDIM / 2, B), B, 0, stream>>>(esrc, edst, logit, den,
                                                             xl, gout);
    add_bc_k<<<cdiv((long)N_NODES * FDIM, B), B, 0, stream>>>(gout, bc);
    // head transform: hn = relu(gout @ Wh + bh)  (WMMA, fused epilogue)
    {
      dim3 grid(cdiv(N_NODES / 16, 8), HID / 16);
      gemm_wmma_bf16<FDIM, true, true><<<grid, 256, 0, stream>>>(gout, Wh, bh, hn,
                                                                 N_NODES, HID);
    }
    bn_k<<<cdiv((long)N_NODES * HID, B), B, 0, stream>>>(hn, gamma, beta, rm, rv);
    // TopK pooling
    pnorm_k<<<1, 64, 0, stream>>>(pv, pnorm);
    score_k<<<cdiv(N_NODES, B), B, 0, stream>>>(hn, pv, pnorm, nmask, batch, vcnt, score);
    kcount_k<<<1, NUM_GRAPHS, 0, stream>>>(vcnt, kk, ratios[L]);
    topk_keep_k<<<cdiv(N_NODES, B), B, 0, stream>>>(score, nmask, batch, starts,
                                                    counts, kk, keep);
    scale_k<<<cdiv((long)N_NODES * HID, B), B, 0, stream>>>(hn, score, keep);
    emask_k<<<cdiv(N_EDGES, B), B, 0, stream>>>(emask, esrc, edst, keep);
    // mean-pool readout into reads[:, L*64 : L*64+64]
    pool_sum_k<<<cdiv((long)N_NODES * HID, B), B, 0, stream>>>(hn, batch, reads, L * HID);
    pool_cnt_k<<<cdiv(N_NODES, B), B, 0, stream>>>(keep, batch, pcnt);
    pool_div_k<<<cdiv(NUM_GRAPHS * HID, B), B, 0, stream>>>(reads, pcnt, L * HID);

    // next layer state
    hin = hn;
    int* t = nmask; nmask = keep; keep = t;
  }

  // final MLP (tiny: 256x192x64 + 256x64x1)
  mlp1_k<<<cdiv(NUM_GRAPHS * HID, B), B, 0, stream>>>(reads, Wf1, bf1, z1);
  mlp2_k<<<1, NUM_GRAPHS, 0, stream>>>(z1, Wf2, bf2, outp);
}
